// MOELinearB_87651692577402
// MI455X (gfx1250) — matrix operand order repaired and checked by reading, not compiled, not run on hardware
//
#include <hip/hip_runtime.h>

typedef __attribute__((ext_vector_type(16))) _Float16 v16h;
typedef __attribute__((ext_vector_type(8)))  _Float16 v8h;
typedef __attribute__((ext_vector_type(4)))  _Float16 v4h;
typedef __attribute__((ext_vector_type(8)))  float    v8f;
typedef __attribute__((ext_vector_type(4)))  float    v4f;

namespace {
constexpr int kE = 8;       // experts
constexpr int kR = 128;     // rank (K)
constexpr int kO = 4096;    // out features (N)
constexpr int kT = 8192;    // tokens (M)
constexpr int BM = 128;     // M tile
constexpr int BN = 128;     // N tile
constexpr int BK = 64;      // K panel staged per round (2 rounds)
constexpr int PITCH = BK + 8; // 72 halfs = 144 B = 9 x 16B -> conflict-free b128 rows
}

__global__ __launch_bounds__(256)
void moe_lorab_wmma_f16(const float* __restrict__ X,
                        const float* __restrict__ W,
                        float* __restrict__ Out)
{
  __shared__ __attribute__((aligned(16))) _Float16 ldsA[BM][PITCH];
  __shared__ __attribute__((aligned(16))) _Float16 ldsB[BN][PITCH];

  const int e    = blockIdx.z;
  const int m0   = blockIdx.y * BM;
  const int n0   = blockIdx.x * BN;
  const int tid  = threadIdx.x;
  const int lane = tid & 31;
  const int wave = tid >> 5;
  const int wm   = (wave & 3) << 5;   // wave M offset: 0,32,64,96
  const int wn   = (wave >> 2) << 6;  // wave N offset: 0,64
  const int lr   = lane & 15;         // row within 16x16 fragment
  const int hs   = lane >> 4;         // lane-half select

  const float* Xp = X + (size_t)m0 * (kE * kR) + e * kR;           // [128, K] panel base
  const float* Wp = W + ((size_t)e * kO + n0) * (size_t)kR;        // [128, K] panel base

  v8f acc[2][4];
  #pragma unroll
  for (int mi = 0; mi < 2; ++mi)
    #pragma unroll
    for (int ni = 0; ni < 4; ++ni)
      acc[mi][ni] = {};

  #pragma unroll
  for (int kp = 0; kp < 2; ++kp) {
    if (kp) __syncthreads();          // protect LDS reuse between panels
    // ---- stage 128x64 f32 panels of X and W into LDS as f16 -----------------
    #pragma unroll
    for (int i = 0; i < 8; ++i) {
      const int idx = tid + (i << 8);       // 0..2047 float4 slots
      const int row = idx >> 4;             // 0..127
      const int c4  = (idx & 15) << 2;      // 0..60 step 4
      v4f xa = *(const v4f*)(Xp + (size_t)row * (kE * kR) + kp * BK + c4);
      v4f wb = *(const v4f*)(Wp + (size_t)row * kR        + kp * BK + c4);
      *(v4h*)&ldsA[row][c4] = __builtin_convertvector(xa, v4h);
      *(v4h*)&ldsB[row][c4] = __builtin_convertvector(wb, v4h);
    }
    __syncthreads();

    // ---- two K=32 WMMA steps per panel --------------------------------------
    #pragma unroll
    for (int ks = 0; ks < 2; ++ks) {
      const int kb = ks << 5;               // 0 or 32 within panel

      v16h afrag[2];
      #pragma unroll
      for (int mi = 0; mi < 2; ++mi) {
        // A 16-bit 16x32 layout: lane<16 holds K 0-7 & 16-23; lane>=16 holds K 8-15 & 24-31
        const _Float16* p = &ldsA[wm + (mi << 4) + lr][kb];
        v8h lo = *(const v8h*)(p + (hs << 3));
        v8h hi = *(const v8h*)(p + 16 + (hs << 3));
        afrag[mi] = __builtin_shufflevector(lo, hi,
            0,1,2,3,4,5,6,7,8,9,10,11,12,13,14,15);
      }

      v16h bfrag[4];
      #pragma unroll
      for (int ni = 0; ni < 4; ++ni) {
        // B (KxN): lane = N; lanes 0-15 hold K 0-7 & 16-23, lanes 16-31 hold K 8-15 & 24-31.
        // W is [O, r] row-major == B^T, so W rows load directly as the B operand.
        const _Float16* p = &ldsB[wn + (ni << 4) + lr][kb + (hs << 3)];
        v8h lo = *(const v8h*)(p);
        v8h hi = *(const v8h*)(p + 16);
        bfrag[ni] = __builtin_shufflevector(lo, hi,
            0,1,2,3,4,5,6,7,8,9,10,11,12,13,14,15);
      }

      #pragma unroll
      for (int mi = 0; mi < 2; ++mi)
        #pragma unroll
        for (int ni = 0; ni < 4; ++ni)
          acc[mi][ni] = __builtin_amdgcn_wmma_f32_16x16x32_f16(
              /*neg_a=*/false, afrag[mi], /*neg_b=*/false, bfrag[ni],
              /*c_mod=*/(short)0, acc[mi][ni],
              /*reuse_a=*/false, /*reuse_b=*/false);
    }
  }

  // ---- write out: Out[t][e][o] f32; C layout: VGPR v -> M=v(+8 upper half), lane%16 -> N
  #pragma unroll
  for (int mi = 0; mi < 2; ++mi) {
    const int mbase = m0 + wm + (mi << 4) + (hs << 3);
    #pragma unroll
    for (int ni = 0; ni < 4; ++ni) {
      const int n = n0 + wn + (ni << 4) + lr;
      #pragma unroll
      for (int v = 0; v < 8; ++v) {
        const size_t m = (size_t)(mbase + v);
        Out[(m * kE + e) * (size_t)kO + n] = acc[mi][ni][v];
      }
    }
  }
}

extern "C" void kernel_launch(void* const* d_in, const int* in_sizes, int n_in,
                              void* d_out, int out_size, void* d_ws, size_t ws_size,
                              hipStream_t stream) {
  const float* X = (const float*)d_in[0];   // [8192, 8*128] f32
  const float* W = (const float*)d_in[1];   // [8, 4096, 128] f32
  float* Out = (float*)d_out;               // [8192, 8, 4096] f32
  (void)in_sizes; (void)n_in; (void)out_size; (void)d_ws; (void)ws_size;

  dim3 grid(kO / BN, kT / BM, kE);          // (32, 64, 8)
  dim3 block(256);
  hipLaunchKernelGGL(moe_lorab_wmma_f16, grid, block, 0, stream, X, W, Out);
}
